// Virtual_Encoder_37383395345197
// MI455X (gfx1250) — matrix operand order repaired and checked by reading, not compiled, not run on hardware
//
#include <hip/hip_runtime.h>
#include <hip/hip_bf16.h>

// ---------------------------------------------------------------------------
// GIN + virtual node encoder for MI455X (gfx1250, wave32, WMMA).
// B=128 graphs, N=512 nodes, EG=8192 edges/graph, CIN=64, H=128, DEPTH=3.
//
// Working set (hA f32 + hZ f32 + hT bf16 ~ 84MB) is L2-resident (192MB);
// the scatter atomics dominate, GEMMs run from L2/LDS via bf16 WMMA.
// ---------------------------------------------------------------------------

#define GB    128          // graphs
#define GN    512          // real nodes per graph
#define GEG   8192         // edges per graph
#define NEDGE (GB * GEG)   // 1,048,576 data edges
#define CIN   64
#define HDIM  128
#define NP1   (GN + 1)     // 513
#define NTOT  (GB * NP1)   // 65,664 nodes incl. virtual

typedef __attribute__((ext_vector_type(16))) __bf16 v16bf;
typedef __attribute__((ext_vector_type(8)))  float  v8f;

// ---------------------------------------------------------------------------
// 1. Edge remap: old id (g*512 + i) -> new id (g*513 + i), int64 -> int32
// ---------------------------------------------------------------------------
__global__ void remap_edges(const long long* __restrict__ src,
                            const long long* __restrict__ dst,
                            int* __restrict__ s2, int* __restrict__ d2, int nE) {
    int e = blockIdx.x * blockDim.x + threadIdx.x;
    if (e >= nE) return;
    long long s = src[e], d = dst[e];
    s2[e] = (int)(s + (s >> 9));   // s / 512
    d2[e] = (int)(d + (d >> 9));
}

// ---------------------------------------------------------------------------
// 2. Pack f32 weight [K,128] into bf16 WMMA B-fragment order:
//    Wp[((c*8 + colTile)*32 + lane)*16 + j] = W[c*32 + kbase + j][colTile*16 + (lane&15)]
//    with kbase = (lane<16) ? 0 : 16.  Each lane's fragment = 32 contiguous bytes.
// ---------------------------------------------------------------------------
__global__ void pack_w(const float* __restrict__ W, __bf16* __restrict__ Wp, int K) {
    int idx = blockIdx.x * blockDim.x + threadIdx.x;
    if (idx >= K * HDIM) return;
    int j    = idx & 15;
    int lane = (idx >> 4) & 31;
    int ct   = (idx >> 9) & 7;
    int c    = idx >> 12;
    int kbase = (lane & 16) ? 16 : 0;
    int k = c * 32 + kbase + j;
    int n = ct * 16 + (lane & 15);
    Wp[idx] = (__bf16)W[(long)k * HDIM + n];
}

// ---------------------------------------------------------------------------
// 3. Build h0: [NTOT, 64], real rows = x, virtual rows = 0
// ---------------------------------------------------------------------------
__global__ void build_h0(const float* __restrict__ x, float* __restrict__ h) {
    int idx = blockIdx.x * blockDim.x + threadIdx.x;     // over NTOT*64
    if (idx >= NTOT * CIN) return;
    int node = idx / CIN, f = idx - node * CIN;
    int g = node / NP1, local = node - g * NP1;
    h[idx] = (local < GN) ? x[((long)g * GN + local) * CIN + f] : 0.0f;
}

// ---------------------------------------------------------------------------
// 4. init_z: z = h + (real node ? h[virtual row of its graph] : 0)
// ---------------------------------------------------------------------------
__global__ void init_z(const float* __restrict__ h, float* __restrict__ z, int F) {
    int idx = blockIdx.x * blockDim.x + threadIdx.x;     // over NTOT*F
    if (idx >= NTOT * F) return;
    int node = idx / F, f = idx - node * F;
    int g = node / NP1, local = node - g * NP1;
    float v = h[idx];
    if (local < GN) v += h[((long)g * NP1 + GN) * F + f];
    z[idx] = v;
}

// ---------------------------------------------------------------------------
// 5. graph_sum: z[virt] += sum over the 512 real rows of the graph
// ---------------------------------------------------------------------------
__global__ void graph_sum(const float* __restrict__ h, float* __restrict__ z, int F) {
    int g = blockIdx.x, f = threadIdx.x;
    const float* base = h + (long)g * NP1 * F + f;
    float s = 0.0f;
    for (int i = 0; i < GN; ++i) s += base[(long)i * F];
    z[((long)g * NP1 + GN) * F + f] += s;
}

// ---------------------------------------------------------------------------
// 6. edge_scatter: z[dst] += h[src] (f32 atomics, float4 gather per thread)
// ---------------------------------------------------------------------------
__global__ void edge_scatter(const float* __restrict__ h, float* __restrict__ z,
                             const int* __restrict__ s2, const int* __restrict__ d2,
                             int F) {
    int tid = blockIdx.x * blockDim.x + threadIdx.x;
    int perEdge = F >> 2;                  // 16 (F=64) or 32 (F=128)
    int e = tid / perEdge;
    if (e >= NEDGE) return;
    int fc = (tid - e * perEdge) << 2;
    const float4 v = *(const float4*)(h + (long)s2[e] * F + fc);
    float* dp = z + (long)d2[e] * F + fc;
    atomicAdd(dp + 0, v.x);
    atomicAdd(dp + 1, v.y);
    atomicAdd(dp + 2, v.z);
    atomicAdd(dp + 3, v.w);
}

// ---------------------------------------------------------------------------
// 7. WMMA GEMM: C[M,128] = act(A[M,K] @ W[K,128] + bias)
//    - K is a template constant (64 or 128) so the chunk loop fully unrolls.
//    - 256 threads = 8 waves; wave w owns the 16x16 tile at cols 16w..16w+15;
//      block b owns rows 16b..16b+15.
//    - A tile (shared by all 8 waves) is staged once in LDS, converted to
//      bf16 and pre-permuted into WMMA A-fragment order, so each lane's
//      per-chunk fragment is one 32-byte ds read (2x ds_load_b128).
//    - W is pre-packed into B-fragment order (2x global_load_b128 per chunk).
//    Fragment orders (A/B use this code's contiguous k, same for both; not the ISA table order):
//      A 16x32: lane L -> row m=L&15, kf=(L<16?0:16);
//               a[j]=A[m][c32+kf+j]
//      B 32x16: lane L -> col n=L&15, kb=(L<16?0:16); b[j]=W[c32+kb+j][n]
//      C/D 16x16 f32: VGPR r -> row m=r+(L<16?0:8), col n=L&15
// ---------------------------------------------------------------------------
template <int K, bool A_BF16, bool OUT_BF16>
__global__ __launch_bounds__(256)
void gemm_tile(const void* __restrict__ Av, const __bf16* __restrict__ Wp,
               const float* __restrict__ bias, void* __restrict__ Cv,
               int reluOut) {
    __shared__ alignas(32) __bf16 As[16 * K];   // A tile in fragment order

    const int tid     = threadIdx.x;
    const int rowBase = blockIdx.x * 16;

    // Cooperative fill of the A tile, permuting into fragment order.
    for (int idx = tid; idx < 16 * K; idx += 256) {
        int j    = idx & 15;
        int lane = (idx >> 4) & 31;
        int c    = idx >> 9;
        int m    = lane & 15;
        int kf   = (lane & 16) ? 16 : 0;
        int k    = c * 32 + kf + j;
        long gidx = (long)(rowBase + m) * K + k;
        if (A_BF16) As[idx] = ((const __bf16*)Av)[gidx];
        else        As[idx] = (__bf16)(((const float*)Av)[gidx]);
    }
    __syncthreads();

    const int lane = tid & 31;
    const int wave = tid >> 5;          // 0..7 -> column tile

    v8f acc = {};
#pragma unroll
    for (int c = 0; c < K / 32; ++c) {
        v16bf a = *(const v16bf*)(As + ((c * 32 + lane) << 4));
        v16bf b = *(const v16bf*)(Wp + ((((c * 8 + wave) * 32) + lane) << 4));
        acc = __builtin_amdgcn_wmma_f32_16x16x32_bf16(
            false, a, false, b, (short)0, acc, false, false);
    }

    const int m  = lane & 15;
    const int n  = wave * 16 + m;
    const int ro = (lane & 16) ? 8 : 0;
    const float bv = bias[n];
#pragma unroll
    for (int r = 0; r < 8; ++r) {
        int row = rowBase + r + ro;
        float v = acc[r] + bv;
        if (reluOut) v = fmaxf(v, 0.0f);
        if (OUT_BF16) ((__bf16*)Cv)[(long)row * HDIM + n] = (__bf16)v;
        else          ((float*)Cv)[(long)row * HDIM + n] = v;
    }
}

// ---------------------------------------------------------------------------
// 8. Gather virtual-node rows with ReLU -> gx [128,128] bf16
// ---------------------------------------------------------------------------
__global__ void gather_virt_relu(const float* __restrict__ h, __bf16* __restrict__ gx) {
    int g = blockIdx.x, f = threadIdx.x;
    float v = h[((long)g * NP1 + GN) * HDIM + f];
    gx[g * HDIM + f] = (__bf16)fmaxf(v, 0.0f);
}

// ---------------------------------------------------------------------------
// Host-side launcher
// ---------------------------------------------------------------------------
extern "C" void kernel_launch(void* const* d_in, const int* in_sizes, int n_in,
                              void* d_out, int out_size, void* d_ws, size_t ws_size,
                              hipStream_t stream) {
    (void)in_sizes; (void)n_in; (void)out_size; (void)ws_size;

    const float*     x     = (const float*)d_in[0];
    const long long* eidx  = (const long long*)d_in[1];      // [2, NEDGE] int64
    const float*     w1_0  = (const float*)d_in[2];
    const float*     b1_0  = (const float*)d_in[3];
    const float*     w2_0  = (const float*)d_in[4];
    const float*     b2_0  = (const float*)d_in[5];
    const float*     w1_l  = (const float*)d_in[6];          // [3,128,128]
    const float*     b1_l  = (const float*)d_in[7];          // [3,128]
    const float*     w2_l  = (const float*)d_in[8];
    const float*     b2_l  = (const float*)d_in[9];
    const float*     lin_w = (const float*)d_in[10];
    const float*     lin_b = (const float*)d_in[11];
    float*           out   = (float*)d_out;

    char* ws = (char*)d_ws;
    size_t off = 0;
    auto carve = [&](size_t bytes) -> char* {
        char* p = ws + off;
        off += (bytes + 255) & ~(size_t)255;
        return p;
    };
    float*  hA    = (float*)carve((size_t)NTOT * HDIM * 4);   // layer activations (f32)
    float*  hZ    = (float*)carve((size_t)NTOT * HDIM * 4);   // z = h + agg      (f32)
    __bf16* hT    = (__bf16*)carve((size_t)NTOT * HDIM * 2);  // t = relu(z@w1+b1) (bf16)
    int*    s2    = (int*)carve((size_t)NEDGE * 4);
    int*    d2    = (int*)carve((size_t)NEDGE * 4);
    __bf16* w10p  = (__bf16*)carve((size_t)CIN * HDIM * 2);
    __bf16* w20p  = (__bf16*)carve((size_t)HDIM * HDIM * 2);
    __bf16* w1lp  = (__bf16*)carve((size_t)3 * HDIM * HDIM * 2);
    __bf16* w2lp  = (__bf16*)carve((size_t)3 * HDIM * HDIM * 2);
    __bf16* linwp = (__bf16*)carve((size_t)HDIM * HDIM * 2);
    __bf16* gx    = (__bf16*)carve((size_t)GB * HDIM * 2);

    const int T = 256;

    // --- preprocessing -----------------------------------------------------
    remap_edges<<<(NEDGE + T - 1) / T, T, 0, stream>>>(eidx, eidx + NEDGE, s2, d2, NEDGE);

    pack_w<<<(CIN * HDIM + T - 1) / T, T, 0, stream>>>(w1_0, w10p, CIN);
    pack_w<<<(HDIM * HDIM + T - 1) / T, T, 0, stream>>>(w2_0, w20p, HDIM);
    for (int k = 0; k < 3; ++k) {
        pack_w<<<(HDIM * HDIM + T - 1) / T, T, 0, stream>>>(
            w1_l + (size_t)k * HDIM * HDIM, w1lp + (size_t)k * HDIM * HDIM, HDIM);
        pack_w<<<(HDIM * HDIM + T - 1) / T, T, 0, stream>>>(
            w2_l + (size_t)k * HDIM * HDIM, w2lp + (size_t)k * HDIM * HDIM, HDIM);
    }
    pack_w<<<(HDIM * HDIM + T - 1) / T, T, 0, stream>>>(lin_w, linwp, HDIM);

    build_h0<<<((size_t)NTOT * CIN + T - 1) / T, T, 0, stream>>>(x, hA);

    const int gemmGrid = NTOT / 16;   // 4104 tiles of 16 rows

    // --- layer 0 (K = CIN = 64), outer ReLU applied after conv -------------
    {
        const int F = CIN;
        init_z<<<((size_t)NTOT * F + T - 1) / T, T, 0, stream>>>(hA, hZ, F);
        graph_sum<<<GB, F, 0, stream>>>(hA, hZ, F);
        edge_scatter<<<((size_t)NEDGE * (F / 4) + T - 1) / T, T, 0, stream>>>(hA, hZ, s2, d2, F);
        gemm_tile<64, false, true><<<gemmGrid, T, 0, stream>>>(hZ, w10p, b1_0, hT, 1);
        gemm_tile<128, true, false><<<gemmGrid, T, 0, stream>>>(hT, w20p, b2_0, hA, 1);
    }

    // --- layers k = 0..2 (K = H = 128); ReLU after conv except last --------
    for (int k = 0; k < 3; ++k) {
        const int F = HDIM;
        init_z<<<((size_t)NTOT * F + T - 1) / T, T, 0, stream>>>(hA, hZ, F);
        graph_sum<<<GB, F, 0, stream>>>(hA, hZ, F);
        edge_scatter<<<((size_t)NEDGE * (F / 4) + T - 1) / T, T, 0, stream>>>(hA, hZ, s2, d2, F);
        gemm_tile<128, false, true><<<gemmGrid, T, 0, stream>>>(
            hZ, w1lp + (size_t)k * HDIM * HDIM, b1_l + k * HDIM, hT, 1);
        gemm_tile<128, true, false><<<gemmGrid, T, 0, stream>>>(
            hT, w2lp + (size_t)k * HDIM * HDIM, b2_l + k * HDIM, hA, (k < 2) ? 1 : 0);
    }

    // --- head: relu(gx) @ lin_w + lin_b ------------------------------------
    gather_virt_relu<<<GB, HDIM, 0, stream>>>(hA, gx);
    gemm_tile<128, true, false><<<GB / 16, T, 0, stream>>>(gx, linwp, lin_b, out, 0);
}
